// AttentionSelf_30030411333841
// MI455X (gfx1250) — matrix operand order repaired and checked by reading, not compiled, not run on hardware
//
#include <hip/hip_runtime.h>

// CDNA5 / gfx1250 self-attention: bf16 WMMA, f32 accumulate, wave-private
// online softmax (no barriers in the main loop), async LDS staging for Q.
typedef __attribute__((ext_vector_type(16))) __bf16 v16bf;
typedef __attribute__((ext_vector_type(8)))  __bf16 v8bf;
typedef __attribute__((ext_vector_type(8)))  float  v8f;
typedef __attribute__((ext_vector_type(4)))  float  v4f;

#define BATCH 8
#define S_LEN 2048
#define D_DIM 512
#define O_DIM 512
#define M_TOT (BATCH * S_LEN)   // 16384 flattened rows

__device__ __forceinline__ v16bf join16(v8bf lo, v8bf hi) {
    v16bf r;
#pragma unroll
    for (int i = 0; i < 8; ++i) { r[i] = lo[i]; r[i + 8] = hi[i]; }
    return r;
}

// A-operand (16x32 bf16) from an f32 row: lane L (m = L%16) holds
// K = {base..base+7, base+16..base+23}, base = (L/16)*8.
__device__ __forceinline__ v16bf load_a_from_f32(const float* __restrict__ row, int base) {
    v4f a0 = *(const v4f*)(row + base);
    v4f a1 = *(const v4f*)(row + base + 4);
    v4f a2 = *(const v4f*)(row + base + 16);
    v4f a3 = *(const v4f*)(row + base + 20);
    v16bf r;
#pragma unroll
    for (int i = 0; i < 4; ++i) {
        r[i]      = (__bf16)a0[i];
        r[4 + i]  = (__bf16)a1[i];
        r[8 + i]  = (__bf16)a2[i];
        r[12 + i] = (__bf16)a3[i];
    }
    return r;
}

// ---------------------------------------------------------------------------
// Kernel 1: QKV projection. Each wave owns a 16x64 tile (4 WMMAs per A-load).
// z=0 -> Q row-major bf16, z=1 -> K row-major bf16, z=2 -> V transposed [B,O,S].
// ---------------------------------------------------------------------------
__global__ __launch_bounds__(256)
void qkv_gemm_kernel(const float* __restrict__ x, const float* __restrict__ w,
                     __bf16* __restrict__ qb, __bf16* __restrict__ kb,
                     __bf16* __restrict__ vt)
{
    __shared__ __align__(16) __bf16 Wt[64][40];   // [n][k], padded rows (80B)

    const int tid  = threadIdx.x;
    const int wave = tid >> 5;
    const int lane = tid & 31;
    const int half = lane >> 4;
    const int l16  = lane & 15;
    const int abase = half * 8;

    int bid = blockIdx.x;
    const int z  = bid % 3;               bid /= 3;
    const int nt = bid % (O_DIM / 64);    bid /= (O_DIM / 64);
    const int ms = bid;                   // 0..127
    const int n0 = nt * 64;
    const int m0 = ms * 128 + wave * 16;

    const float* W = w + (size_t)z * D_DIM * O_DIM;

    v8f acc[4];
#pragma unroll
    for (int j = 0; j < 4; ++j) acc[j] = (v8f)0.f;

    for (int k0 = 0; k0 < D_DIM; k0 += 32) {
        __syncthreads();
        // Stage W[k0..k0+32, n0..n0+64] transposed: 2048 elems = 512 v4f, 2/thread.
#pragma unroll
        for (int i = 0; i < 2; ++i) {
            const int idx = tid + i * 256;          // 0..511
            const int kk  = idx >> 4;               // 0..31
            const int nn4 = (idx & 15) * 4;         // 0..60
            v4f wv = *(const v4f*)&W[(size_t)(k0 + kk) * O_DIM + n0 + nn4];
#pragma unroll
            for (int e = 0; e < 4; ++e) Wt[nn4 + e][kk] = (__bf16)wv[e];
        }
        __syncthreads();

        const float* arow = x + (size_t)(m0 + l16) * D_DIM + k0;
        v16bf A = load_a_from_f32(arow, abase);
#pragma unroll
        for (int j = 0; j < 4; ++j) {
            v8bf b0 = *(const v8bf*)&Wt[j * 16 + l16][half * 8];
            v8bf b1 = *(const v8bf*)&Wt[j * 16 + l16][half * 8 + 16];
            v16bf Bm = join16(b0, b1);
            acc[j] = __builtin_amdgcn_wmma_f32_16x16x32_bf16(
                         false, A, false, Bm, (short)0, acc[j], false, false);
        }
    }

    // C/D layout: VGPR v -> row m = v + 8*half, col n = l16 (+16*j).
    const int rbase = m0 + half * 8;
    if (z == 2) {
        const int b = rbase / S_LEN;
        const int s = rbase % S_LEN;
#pragma unroll
        for (int j = 0; j < 4; ++j) {
            __bf16* dst = vt + ((size_t)b * O_DIM + (n0 + j * 16 + l16)) * S_LEN + s;
            v8bf pk;
#pragma unroll
            for (int v = 0; v < 8; ++v) pk[v] = (__bf16)acc[j][v];
            *(v8bf*)dst = pk;
        }
    } else {
        __bf16* dst = (z == 0 ? qb : kb);
#pragma unroll
        for (int j = 0; j < 4; ++j)
#pragma unroll
            for (int v = 0; v < 8; ++v)
                dst[(size_t)(rbase + v) * O_DIM + n0 + j * 16 + l16] = (__bf16)acc[j][v];
    }
}

// ---------------------------------------------------------------------------
// Kernel 2: flash attention, barrier-free main loop.
// WG = 128 thr = 4 waves covering 32 query rows: wave = (row-group g, o-half h).
// Each wave: full 512-deep scores for its 16 rows (x2 redundancy across the
// o-half pair), register-resident online softmax via __shfl_xor row
// reductions, LDS transpose of P (wave-private, s_wait_dscnt only), then
// P @ V over its 256-wide o-slice. 48 WMMAs / 32 keys / wave, no barriers.
// ---------------------------------------------------------------------------
__global__ __launch_bounds__(128)
void attn_kernel(const __bf16* __restrict__ qb, const __bf16* __restrict__ kb,
                 const __bf16* __restrict__ vt, float* __restrict__ out)
{
    __shared__ __align__(16) __bf16 Qs[2][16][520];   // padded rows: bank-safe
    __shared__ __align__(16) __bf16 Pb[4][16][40];    // per-wave P transpose buf

    const int tid  = threadIdx.x;
    const int wave = tid >> 5;
    const int lane = tid & 31;
    const int half = lane >> 4;
    const int l16  = lane & 15;
    const int abase = half * 8;

    const int g  = wave >> 1;            // row-group 0/1
    const int h  = wave & 1;             // o-half 0/1
    const int ow = h * 256;

    const int b   = blockIdx.x >> 6;     // / (S_LEN/32)
    const int qt  = blockIdx.x & 63;
    const size_t row0 = (size_t)b * S_LEN + qt * 32;   // 32 rows per WG

    // ---- Stage 32x512 Q rows into LDS with gfx1250 async LDS loads ----
    {
        __bf16* qsp = &Qs[0][0][0];
#pragma unroll
        for (int i = 0; i < 16; ++i) {
            const int ch = tid + i * 128;            // 0..2047 : 16B chunks
            const int r  = ch >> 6;                  // row 0..31
            const int c8 = (ch & 63) * 8;            // col 0..504
            const unsigned lds_off =
                (unsigned)(uintptr_t)(qsp + (size_t)r * 520 + c8);
            const unsigned long long ga =
                (unsigned long long)(uintptr_t)(qb + (row0 + r) * O_DIM + c8);
            asm volatile("global_load_async_to_lds_b128 %0, %1, off"
                         :: "v"(lds_off), "v"(ga) : "memory");
        }
        asm volatile("s_wait_asynccnt 0x0" ::: "memory");
    }
    __syncthreads();

    v8f acc[16];
#pragma unroll
    for (int j = 0; j < 16; ++j) acc[j] = (v8f)0.f;
    float m_i[8], l_i[8];
#pragma unroll
    for (int v = 0; v < 8; ++v) { m_i[v] = -1e30f; l_i[v] = 0.f; }

    const size_t bS = (size_t)b * S_LEN;

    for (int kb0 = 0; kb0 < S_LEN; kb0 += 32) {
        // --- scores: two 16x16 tiles, K = 512 reduction from LDS Q ---
        v8f sc0 = (v8f)0.f, sc1 = (v8f)0.f;
        const __bf16* kp0 = kb + (bS + kb0 + l16) * O_DIM + half * 16;
        const __bf16* kp1 = kp0 + (size_t)16 * O_DIM;
#pragma unroll 8
        for (int oc = 0; oc < 16; ++oc) {
            v8bf a0 = *(const v8bf*)&Qs[g][l16][oc * 32 + half * 16];
            v8bf a1 = *(const v8bf*)&Qs[g][l16][oc * 32 + half * 16 + 8];
            v16bf A = join16(a0, a1);
            v16bf B0 = join16(*(const v8bf*)(kp0 + oc * 32),
                              *(const v8bf*)(kp0 + oc * 32 + 8));
            sc0 = __builtin_amdgcn_wmma_f32_16x16x32_bf16(
                      false, A, false, B0, (short)0, sc0, false, false);
            v16bf B1 = join16(*(const v8bf*)(kp1 + oc * 32),
                              *(const v8bf*)(kp1 + oc * 32 + 8));
            sc1 = __builtin_amdgcn_wmma_f32_16x16x32_bf16(
                      false, A, false, B1, (short)0, sc1, false, false);
        }

        // --- wave-private online softmax (rows live in VGPR v + lane group) ---
        float cr[8];
#pragma unroll
        for (int v = 0; v < 8; ++v) {
            float s0 = sc0[v] * 0.125f;              // 1/sqrt(64)
            float s1 = sc1[v] * 0.125f;
            float t = fmaxf(s0, s1);                 // rowmax over 16 lanes:
            t = fmaxf(t, __shfl_xor(t, 1));
            t = fmaxf(t, __shfl_xor(t, 2));
            t = fmaxf(t, __shfl_xor(t, 4));
            t = fmaxf(t, __shfl_xor(t, 8));
            const float mnew = fmaxf(m_i[v], t);
            const float corr = __expf(m_i[v] - mnew);
            m_i[v] = mnew;
            cr[v]  = corr;
            const float p0 = __expf(s0 - mnew);
            const float p1 = __expf(s1 - mnew);
            Pb[wave][v + 8 * half][l16]      = (__bf16)p0;
            Pb[wave][v + 8 * half][16 + l16] = (__bf16)p1;
            float rs = p0 + p1;
            rs += __shfl_xor(rs, 1);
            rs += __shfl_xor(rs, 2);
            rs += __shfl_xor(rs, 4);
            rs += __shfl_xor(rs, 8);
            l_i[v] = l_i[v] * corr + rs;
        }
        asm volatile("s_wait_dscnt 0x0" ::: "memory");   // wave-private LDS RAW

        // --- read P back in A-operand layout, accumulate P @ V ---
        v8bf p0 = *(const v8bf*)&Pb[wave][l16][half * 16];
        v8bf p1 = *(const v8bf*)&Pb[wave][l16][half * 16 + 8];
        v16bf Ap = join16(p0, p1);
#pragma unroll 8
        for (int j = 0; j < 16; ++j) {
            const __bf16* vp = vt + ((size_t)b * O_DIM + (ow + j * 16 + l16)) * S_LEN
                                  + kb0 + half * 16;
            v16bf Bv = join16(*(const v8bf*)vp, *(const v8bf*)(vp + 8));
#pragma unroll
            for (int v = 0; v < 8; ++v) acc[j][v] *= cr[v];
            acc[j] = __builtin_amdgcn_wmma_f32_16x16x32_bf16(
                         false, Ap, false, Bv, (short)0, acc[j], false, false);
        }
    }

    // --- epilogue: 1/l scaling, f32 stores ---
    const size_t qrow0 = row0 + g * 16;
    float lv[8];
#pragma unroll
    for (int v = 0; v < 8; ++v) lv[v] = 1.f / l_i[v];
#pragma unroll
    for (int j = 0; j < 16; ++j)
#pragma unroll
        for (int v = 0; v < 8; ++v)
            out[(qrow0 + v + half * 8) * O_DIM + ow + j * 16 + l16] = acc[j][v] * lv[v];
}

// ---------------------------------------------------------------------------
extern "C" void kernel_launch(void* const* d_in, const int* in_sizes, int n_in,
                              void* d_out, int out_size, void* d_ws, size_t ws_size,
                              hipStream_t stream)
{
    const float* x = (const float*)d_in[0];     // [8,2048,512] f32
    const float* w = (const float*)d_in[1];     // [3,512,512]  f32
    float* out = (float*)d_out;                 // [8,2048,512] f32

    __bf16* qb = (__bf16*)d_ws;
    __bf16* kb = qb + (size_t)M_TOT * O_DIM;
    __bf16* vt = kb + (size_t)M_TOT * O_DIM;    // 48 MB workspace

    const int gemm_blocks = (M_TOT / 128) * (O_DIM / 64) * 3;   // 3072
    qkv_gemm_kernel<<<gemm_blocks, 256, 0, stream>>>(x, w, qb, kb, vt);

    const int attn_blocks = BATCH * (S_LEN / 32);               // 512
    attn_kernel<<<attn_blocks, 128, 0, stream>>>(qb, kb, vt, out);
}
